// EncoderBlock_88356067213968
// MI455X (gfx1250) — matrix operand order repaired and checked by reading, not compiled, not run on hardware
//
#include <hip/hip_runtime.h>
#include <hip/hip_bf16.h>

typedef __attribute__((ext_vector_type(16))) _Float16 v16h;
typedef __attribute__((ext_vector_type(8)))  _Float16 v8h;
typedef __attribute__((ext_vector_type(4)))  _Float16 v4h;
typedef __attribute__((ext_vector_type(8)))  float    v8f;
typedef __attribute__((ext_vector_type(4)))  float    v4f;

namespace {
constexpr int B_   = 4;
constexpr int C_   = 64;
constexpr int H_   = 56;
constexpr int W_   = 56;
constexpr int HW_  = H_ * W_;       // 3136
constexpr int L_   = HW_;           // 3136
constexpr int BL_  = B_ * L_;       // 12544  (== 98 * 128, so M % BM == 0 always)
constexpr int E_   = 128;           // d_inner
constexpr int TWOE_= 256;
constexpr int NS_  = 16;            // D_STATE
constexpr int RK_  = 4;             // DT_RANK
constexpr int CO_  = 128;
constexpr int KC3_ = C_ * 9;        // 576
constexpr int DBLW_= RK_ + 2 * NS_; // 36
}

// ---------------------------------------------------------------------------
// Weight pre-conversion f32 -> f16 (weights reused across K-loops; convert once)
// ---------------------------------------------------------------------------
__global__ void f32_to_f16_kernel(const float* __restrict__ s, _Float16* __restrict__ d, int n)
{
    int i = blockIdx.x * blockDim.x + threadIdx.x;
    if (i < n) d[i] = (_Float16)s[i];
}

// ---------------------------------------------------------------------------
// Axial depthwise: out = x + dw7x1(x)+bh + dw1x7(x)+bw   (SAME padding)
// ---------------------------------------------------------------------------
__global__ void axial_kernel(const float* __restrict__ x,
                             const float* __restrict__ wh, const float* __restrict__ bh,
                             const float* __restrict__ ww, const float* __restrict__ bw,
                             float* __restrict__ out)
{
    int idx = blockIdx.x * blockDim.x + threadIdx.x;
    if (idx >= B_ * C_ * HW_) return;
    int w  = idx % W_;
    int h  = (idx / W_) % H_;
    int bc = idx / HW_;
    int c  = bc % C_;
    const float* p = x + (long)bc * HW_;
    float acc = p[h * W_ + w] + bh[c] + bw[c];
#pragma unroll
    for (int t = 0; t < 7; ++t) {
        int hh = h + t - 3;
        if (hh >= 0 && hh < H_) acc += wh[c * 7 + t] * p[hh * W_ + w];
    }
#pragma unroll
    for (int t = 0; t < 7; ++t) {
        int wi = w + t - 3;
        if (wi >= 0 && wi < W_) acc += ww[c * 7 + t] * p[h * W_ + wi];
    }
    out[idx] = acc;
}

// ---------------------------------------------------------------------------
// BatchNorm batch statistics: one block per channel. stats[c]=mean, stats[64+c]=rstd
// ---------------------------------------------------------------------------
__global__ __launch_bounds__(256) void bn_stats_kernel(const float* __restrict__ x,
                                                       float* __restrict__ stats)
{
    int c = blockIdx.x;
    int tid = threadIdx.x;
    float s = 0.f, s2 = 0.f;
    for (int i = tid; i < B_ * HW_; i += 256) {
        int bb = i / HW_;
        int l  = i - bb * HW_;
        float v = x[((long)(bb * C_ + c)) * HW_ + l];
        s += v; s2 += v * v;
    }
    __shared__ float sh0[256], sh1[256];
    sh0[tid] = s; sh1[tid] = s2;
    __syncthreads();
    for (int off = 128; off > 0; off >>= 1) {
        if (tid < off) { sh0[tid] += sh0[tid + off]; sh1[tid] += sh1[tid + off]; }
        __syncthreads();
    }
    if (tid == 0) {
        float m = sh0[0] / (float)(B_ * HW_);
        float v = sh1[0] / (float)(B_ * HW_) - m * m;
        stats[c]      = m;
        stats[C_ + c] = rsqrtf(v + 1e-5f);
    }
}

__global__ void bn_apply_kernel(const float* __restrict__ x, const float* __restrict__ stats,
                                const float* __restrict__ g, const float* __restrict__ bvec,
                                const float* __restrict__ resid, float* __restrict__ out)
{
    int idx = blockIdx.x * blockDim.x + threadIdx.x;
    if (idx >= B_ * C_ * HW_) return;
    int c = (idx / HW_) % C_;
    float v = g[c] * (x[idx] - stats[c]) * stats[C_ + c] + bvec[c];
    if (resid) v += resid[idx];
    out[idx] = v;
}

// ---------------------------------------------------------------------------
// LayerNorm over C per (b,l); input NCHW "skip", output (B*L, C) row-major
// ---------------------------------------------------------------------------
__global__ void layernorm_kernel(const float* __restrict__ skip,
                                 const float* __restrict__ g, const float* __restrict__ b,
                                 float* __restrict__ out)
{
    int idx = blockIdx.x * blockDim.x + threadIdx.x;
    if (idx >= BL_) return;
    int bb = idx / L_;
    int l  = idx - bb * L_;
    const float* base = skip + ((long)bb * C_) * HW_ + l;
    float s = 0.f, s2 = 0.f;
#pragma unroll 8
    for (int c = 0; c < C_; ++c) { float v = base[c * HW_]; s += v; s2 += v * v; }
    float mean = s / (float)C_;
    float var  = s2 / (float)C_ - mean * mean;
    float rstd = rsqrtf(var + 1e-5f);
    float* o = out + (long)idx * C_;
#pragma unroll 8
    for (int c = 0; c < C_; ++c) o[c] = g[c] * (base[c * HW_] - mean) * rstd + b[c];
}

// ---------------------------------------------------------------------------
// Generic tiled WMMA GEMM:  Out[M,N] = act( A[M,K](f32) * Wt[N,K](f16)^T + bias )
//   REQUIRES M % 128 == 0 (true for all call sites: M = 12544).
//   ASRC: 0 = A row-major (lda), 1 = im2col 3x3 SAME from NCHW tensor (K=C*9)
//   EPI : 0 = Out[m*ldc+n], 1 = NCHW transposed store Out[(b*ldc+n)*HW + l]
//   ACT : 0 = none, 1 = softplus
// Tiles BM=128 BN=64 BK=32; 8 waves (4x2), each wave 32x32 via 2x2 WMMA frags.
// Main loop handles only full K-chunks: B tile staged with
// global_load_async_to_lds_b128 (ASYNCcnt), A tile via float4 loads + cvt.
// A single optional tail iteration (K % 32 != 0 -> only dt_proj, K=4) uses
// scalar guarded staging, keeping the hot loop branch-free.
// ---------------------------------------------------------------------------
template <int ASRC, int EPI, int ACT>
__global__ __launch_bounds__(256) void gemm_wmma_kernel(
    const float* __restrict__ A, const _Float16* __restrict__ Wt,
    const float* __restrict__ bias, float* __restrict__ Out,
    int M, int N, int K, int lda, int ldc)
{
    constexpr int BM = 128, BN = 64, BK = 32;
    __shared__ _Float16 sA[BM * BK];   // 8 KB
    __shared__ _Float16 sB[BN * BK];   // 4 KB
    const int tid  = threadIdx.x;
    const int lane = tid & 31;
    const int wid  = tid >> 5;
    const int m0   = blockIdx.y * BM;
    const int n0   = blockIdx.x * BN;
    const int wm   = (wid >> 1) * 32;
    const int wn   = (wid & 1) * 32;
    const int lr   = lane & 15;
    const bool hi  = lane >= 16;

    v8f acc[2][2] = {};

    // One 16x16x32 MMA step per (im,in) pair from staged LDS tiles.
    auto mma_step = [&]() {
        v16h af[2], bf[2];
#pragma unroll
        for (int im = 0; im < 2; ++im) {
            // A 16x32 f16: lanes0-15 K={0..7,16..23}; lanes16-31 K={8..15,24..31}
            const _Float16* pa = &sA[(wm + im * 16 + lr) * BK + (hi ? 8 : 0)];
            v8h a0 = *(const v8h*)pa;
            v8h a1 = *(const v8h*)(pa + 16);
            af[im] = __builtin_shufflevector(a0, a1, 0,1,2,3,4,5,6,7,8,9,10,11,12,13,14,15);
        }
#pragma unroll
        for (int in = 0; in < 2; ++in) {
            // B 32x16 f16: lanes0-15 K={0..7,16..23}; lanes16-31 K={8..15,24..31} (N = lane&15)
            const _Float16* pb = &sB[(wn + in * 16 + lr) * BK + (hi ? 8 : 0)];
            v8h b0 = *(const v8h*)pb;
            v8h b1 = *(const v8h*)(pb + 16);
            bf[in] = __builtin_shufflevector(b0, b1, 0,1,2,3,4,5,6,7,8,9,10,11,12,13,14,15);
        }
#pragma unroll
        for (int im = 0; im < 2; ++im)
#pragma unroll
            for (int in = 0; in < 2; ++in)
                acc[im][in] = __builtin_amdgcn_wmma_f32_16x16x32_f16(
                    false, af[im], false, bf[in], (short)0, acc[im][in], false, false);
    };

    // ---------------- main loop: full 32-wide K chunks ----------------
    const int nkFull = K / BK;
    // B-tile async DMA source: 256 lanes x one 16B chunk (row r, halves c8..c8+7)
    const int br  = tid >> 2;
    const int bc8 = (tid & 3) << 3;
    const int bgn = min(n0 + br, N - 1);          // clamp: OOB cols never stored
    const _Float16* bsrc_row = Wt + (long)bgn * K + bc8;
    const unsigned  bdst = (unsigned)(unsigned long long)(size_t)(&sB[br * BK + bc8]);

    for (int kt = 0; kt < nkFull; ++kt) {
        const int k0 = kt * BK;

        // B tile: DMA straight into LDS, bypassing VGPRs (ASYNCcnt-tracked).
        asm volatile("global_load_async_to_lds_b128 %0, %1, off"
                     :: "v"(bdst), "v"((unsigned long long)(size_t)(bsrc_row + k0))
                     : "memory");

        // A tile: 128 rows x 32 f32 -> f16; batch loads, then cvt+packed stores.
        v4f areg[4];
#pragma unroll
        for (int i = 0; i < 4; ++i) {
            int vidx = i * 256 + tid;          // float4 index in tile (1024 total)
            int r = vidx >> 3, c4 = (vidx & 7) << 2;
            int gm = m0 + r, gk = k0 + c4;     // gm < M guaranteed (M % 128 == 0)
            if (ASRC == 0) {
                areg[i] = *(const v4f*)(A + (long)gm * lda + gk);
            } else {
                int b = gm / HW_, l = gm - b * HW_;
                int h = l / W_,   w = l - h * W_;
                v4f v = {0.f, 0.f, 0.f, 0.f};
#pragma unroll
                for (int j = 0; j < 4; ++j) {
                    int kk = gk + j;
                    int cc = kk / 9, rem = kk - cc * 9;
                    int rr = rem / 3, ss = rem - rr * 3;
                    int ih = h + rr - 1, iw = w + ss - 1;
                    if ((unsigned)ih < (unsigned)H_ && (unsigned)iw < (unsigned)W_)
                        v[j] = A[((long)(b * C_ + cc)) * HW_ + ih * W_ + iw];
                }
                areg[i] = v;
            }
        }
#pragma unroll
        for (int i = 0; i < 4; ++i) {
            int vidx = i * 256 + tid;
            v4h hv;
#pragma unroll
            for (int j = 0; j < 4; ++j) hv[j] = (_Float16)areg[i][j];
            *(v4h*)&sA[vidx << 2] = hv;        // 8B packed ds_store
        }

        asm volatile("s_wait_asynccnt 0x0" ::: "memory");
        __syncthreads();
        mma_step();
        __syncthreads();
    }

    // ---------------- optional tail (K % 32 != 0; only dt_proj K=4) ----------
    if (K & (BK - 1)) {
        const int k0 = nkFull * BK;
#pragma unroll
        for (int i = 0; i < 8; ++i) {
            int idx = i * 256 + tid;
            int r = idx >> 5, cidx = idx & 31;
            int gn = n0 + r, gk = k0 + cidx;
            sB[idx] = (gn < N && gk < K) ? Wt[(long)gn * K + gk] : (_Float16)0.f;
        }
#pragma unroll
        for (int i = 0; i < 16; ++i) {
            int idx = i * 256 + tid;
            int r = idx >> 5, cidx = idx & 31;
            int gm = m0 + r, gk = k0 + cidx;
            float v = 0.f;
            if (ASRC == 0 && gk < K) v = A[(long)gm * lda + gk];
            sA[idx] = (_Float16)v;
        }
        __syncthreads();
        mma_step();
        __syncthreads();
    }

    // ---- epilogue: C/D 16x16 f32: VGPR r -> M=r (lanes0-15) / 8+r (lanes16-31)
#pragma unroll
    for (int im = 0; im < 2; ++im)
#pragma unroll
        for (int in = 0; in < 2; ++in) {
            int gm_base = m0 + wm + im * 16 + (hi ? 8 : 0);
            int gn = n0 + wn + in * 16 + lr;
            if (gn >= N) continue;
#pragma unroll
            for (int r = 0; r < 8; ++r) {
                int gm = gm_base + r;
                float v = acc[im][in][r];
                if (bias) v += bias[gn];
                if (ACT == 1) v = (v > 20.f) ? v : log1pf(__expf(v));
                if (EPI == 0) {
                    Out[(long)gm * ldc + gn] = v;
                } else {
                    int b = gm / HW_, l = gm - b * HW_;
                    Out[((long)(b * ldc + gn)) * HW_ + l] = v;
                }
            }
        }
}

// ---------------------------------------------------------------------------
// Causal depthwise conv1d (k=4, left-pad 3) over L + bias + SiLU.
// ---------------------------------------------------------------------------
__global__ void dwconv_silu_kernel(const float* __restrict__ xz,
                                   const float* __restrict__ cw, const float* __restrict__ cb,
                                   float* __restrict__ xm)
{
    int idx = blockIdx.x * blockDim.x + threadIdx.x;
    if (idx >= BL_ * E_) return;
    int e  = idx % E_;
    int bl = idx / E_;
    int b  = bl / L_;
    int l  = bl - b * L_;
    float acc = cb[e];
#pragma unroll
    for (int j = 0; j < 4; ++j) {
        int ls = l - 3 + j;
        if (ls >= 0) acc += cw[e * 4 + j] * xz[((long)(b * L_ + ls)) * TWOE_ + e];
    }
    xm[idx] = acc / (1.f + __expf(-acc));
}

// ---------------------------------------------------------------------------
// Selective scan: one (b,e) recurrence per 16-lane half-wave; lane = state n.
// ---------------------------------------------------------------------------
__global__ __launch_bounds__(256) void scan_kernel(
    const float* __restrict__ dt, const float* __restrict__ dbl,
    const float* __restrict__ xm, const float* __restrict__ A_log,
    const float* __restrict__ Dp, const float* __restrict__ xz,
    float* __restrict__ yout)
{
    int gt   = blockIdx.x * blockDim.x + threadIdx.x;
    int pair = gt >> 4;          // (b,e) id: 0..511
    int n    = gt & 15;
    int b    = pair >> 7;
    int e    = pair & 127;
    float Ae = -__expf(A_log[e * NS_ + n]);
    float De = Dp[e];
    const float* dtp = dt  + (long)b * L_ * E_    + e;
    const float* Bp  = dbl + (long)b * L_ * DBLW_ + RK_ + n;
    const float* Cp  = dbl + (long)b * L_ * DBLW_ + RK_ + NS_ + n;
    const float* xp  = xm  + (long)b * L_ * E_    + e;
    const float* zp  = xz  + (long)b * L_ * TWOE_ + E_ + e;
    float*       yp  = yout+ (long)b * L_ * E_    + e;
    float h = 0.f;
    for (int l = 0; l < L_; ++l) {
        if (l + 16 < L_) __builtin_prefetch(xp + (long)(l + 16) * E_, 0, 0);
        float dtv = dtp[(long)l * E_];
        float xv  = xp[(long)l * E_];
        h = __expf(dtv * Ae) * h + dtv * Bp[(long)l * DBLW_] * xv;
        float y = h * Cp[(long)l * DBLW_];
        y += __shfl_xor(y, 8, 16);
        y += __shfl_xor(y, 4, 16);
        y += __shfl_xor(y, 2, 16);
        y += __shfl_xor(y, 1, 16);
        if (n == 0) {
            float z  = zp[(long)l * TWOE_];
            float yo = (y + De * xv) * (z / (1.f + __expf(-z)));
            yp[(long)l * E_] = yo;
        }
    }
}

// ---------------------------------------------------------------------------
// Softmax gate over {conv_path, mamba_path}
// ---------------------------------------------------------------------------
__global__ void gate_kernel(const float* __restrict__ conv_path,
                            const float* __restrict__ mamba,
                            const float* __restrict__ gw, const float* __restrict__ gb,
                            float* __restrict__ out)
{
    int idx = blockIdx.x * blockDim.x + threadIdx.x;
    if (idx >= B_ * HW_) return;
    int bb = idx / HW_;
    int l  = idx - bb * HW_;
    const float* cp = conv_path + (long)bb * C_ * HW_ + l;
    const float* mp = mamba     + (long)bb * C_ * HW_ + l;
    float l0 = gb[0], l1 = gb[1];
#pragma unroll 8
    for (int c = 0; c < C_; ++c) {
        float cv = cp[c * HW_], mv = mp[c * HW_];
        l0 += gw[c] * cv        + gw[C_ + c] * mv;
        l1 += gw[2*C_ + c] * cv + gw[3*C_ + c] * mv;
    }
    float mx = fmaxf(l0, l1);
    float e0 = __expf(l0 - mx), e1 = __expf(l1 - mx);
    float g0 = e0 / (e0 + e1), g1 = 1.f - g0;
    float* op = out + (long)bb * C_ * HW_ + l;
#pragma unroll 8
    for (int c = 0; c < C_; ++c) op[c * HW_] = g0 * cp[c * HW_] + g1 * mp[c * HW_];
}

// ---------------------------------------------------------------------------
// 2x2 max-pool stride 2 (VALID)
// ---------------------------------------------------------------------------
__global__ void maxpool_kernel(const float* __restrict__ y, float* __restrict__ out)
{
    int idx = blockIdx.x * blockDim.x + threadIdx.x;
    if (idx >= B_ * CO_ * 28 * 28) return;
    int ow = idx % 28;
    int oh = (idx / 28) % 28;
    int bc = idx / (28 * 28);
    const float* p = y + (long)bc * HW_ + (oh * 2) * W_ + ow * 2;
    out[idx] = fmaxf(fmaxf(p[0], p[1]), fmaxf(p[W_], p[W_ + 1]));
}

// ===========================================================================
extern "C" void kernel_launch(void* const* d_in, const int* in_sizes, int n_in,
                              void* d_out, int out_size, void* d_ws, size_t ws_size,
                              hipStream_t stream)
{
    (void)in_sizes; (void)n_in; (void)out_size; (void)ws_size;
    const float* x         = (const float*)d_in[0];
    const float* dw_h_w    = (const float*)d_in[1];
    const float* dw_h_b    = (const float*)d_in[2];
    const float* dw_w_w    = (const float*)d_in[3];
    const float* dw_w_b    = (const float*)d_in[4];
    const float* bn1_g     = (const float*)d_in[5];
    const float* bn1_b     = (const float*)d_in[6];
    const float* ax_h_w    = (const float*)d_in[7];
    const float* ax_h_b    = (const float*)d_in[8];
    const float* ax_w_w    = (const float*)d_in[9];
    const float* ax_w_b    = (const float*)d_in[10];
    const float* ln_g      = (const float*)d_in[11];
    const float* ln_b      = (const float*)d_in[12];
    const float* in_proj_w = (const float*)d_in[13];
    const float* conv1d_w  = (const float*)d_in[14];
    const float* conv1d_b  = (const float*)d_in[15];
    const float* x_proj_w  = (const float*)d_in[16];
    const float* dt_proj_w = (const float*)d_in[17];
    const float* dt_proj_b = (const float*)d_in[18];
    const float* A_log     = (const float*)d_in[19];
    const float* D_param   = (const float*)d_in[20];
    const float* out_proj_w= (const float*)d_in[21];
    const float* gate_w    = (const float*)d_in[22];
    const float* gate_b    = (const float*)d_in[23];
    const float* bn2_g     = (const float*)d_in[24];
    const float* bn2_b     = (const float*)d_in[25];
    const float* conv3_w   = (const float*)d_in[26];
    const float* conv3_b   = (const float*)d_in[27];

    float* outp = (float*)d_out;
    float* skip = outp + (size_t)B_ * CO_ * 28 * 28;

    float* ws   = (float*)d_ws;
    float* buf0 = ws;                 // 802816  : axial1 / ln_out / gated
    float* buf1 = ws +   802816;      // 802816  : conv_path / amb
    float* buf2 = ws +  1605632;      // 3211264 : xz / conv3_out
    float* buf3 = ws +  4816896;      // 1605632 : xm
    float* buf4 = ws +  6422528;      // 451584  : dbl
    float* buf5 = ws +  6874112;      // 1605632 : dt
    float* buf6 = ws +  8479744;      // 1605632 : y (scan out)
    float* buf7 = ws + 10085376;      // 802816  : mamba_path (NCHW)
    float* stats= ws + 10888192;      // 128     : BN mean/rstd

    // f16 weight pool (halves), after the f32 region
    _Float16* hpool   = (_Float16*)(ws + 10888320);
    _Float16* h_inprj = hpool;            // 16384
    _Float16* h_xprj  = hpool + 16384;    // 4608
    _Float16* h_dtprj = hpool + 20992;    // 512
    _Float16* h_outprj= hpool + 21504;    // 8192
    _Float16* h_conv3 = hpool + 29696;    // 73728

    const int NT = 256;
    const int NPIX = B_ * C_ * HW_;

    // 0) weights -> f16
    f32_to_f16_kernel<<<(16384 + NT - 1) / NT, NT, 0, stream>>>(in_proj_w,  h_inprj, 16384);
    f32_to_f16_kernel<<<( 4608 + NT - 1) / NT, NT, 0, stream>>>(x_proj_w,   h_xprj,   4608);
    f32_to_f16_kernel<<<(  512 + NT - 1) / NT, NT, 0, stream>>>(dt_proj_w,  h_dtprj,   512);
    f32_to_f16_kernel<<<( 8192 + NT - 1) / NT, NT, 0, stream>>>(out_proj_w, h_outprj, 8192);
    f32_to_f16_kernel<<<(73728 + NT - 1) / NT, NT, 0, stream>>>(conv3_w,    h_conv3, 73728);

    // 1) skip = BN1(axial(x))
    axial_kernel<<<(NPIX + NT - 1) / NT, NT, 0, stream>>>(x, dw_h_w, dw_h_b, dw_w_w, dw_w_b, buf0);
    bn_stats_kernel<<<C_, NT, 0, stream>>>(buf0, stats);
    bn_apply_kernel<<<(NPIX + NT - 1) / NT, NT, 0, stream>>>(buf0, stats, bn1_g, bn1_b, nullptr, skip);

    // 2) conv_path = axial(skip)
    axial_kernel<<<(NPIX + NT - 1) / NT, NT, 0, stream>>>(skip, ax_h_w, ax_h_b, ax_w_w, ax_w_b, buf1);

    // 3) Mamba path
    layernorm_kernel<<<(BL_ + NT - 1) / NT, NT, 0, stream>>>(skip, ln_g, ln_b, buf0);
    {
        dim3 g((TWOE_ + 63) / 64, BL_ / 128);
        gemm_wmma_kernel<0, 0, 0><<<g, NT, 0, stream>>>(buf0, h_inprj, nullptr, buf2,
                                                        BL_, TWOE_, C_, C_, TWOE_);
    }
    dwconv_silu_kernel<<<(BL_ * E_ + NT - 1) / NT, NT, 0, stream>>>(buf2, conv1d_w, conv1d_b, buf3);
    {
        dim3 g((DBLW_ + 63) / 64, BL_ / 128);
        gemm_wmma_kernel<0, 0, 0><<<g, NT, 0, stream>>>(buf3, h_xprj, nullptr, buf4,
                                                        BL_, DBLW_, E_, E_, DBLW_);
    }
    {
        dim3 g((E_ + 63) / 64, BL_ / 128);
        gemm_wmma_kernel<0, 0, 1><<<g, NT, 0, stream>>>(buf4, h_dtprj, dt_proj_b, buf5,
                                                        BL_, E_, RK_, DBLW_, E_);
    }
    scan_kernel<<<32, NT, 0, stream>>>(buf5, buf4, buf3, A_log, D_param, buf2, buf6);
    {
        dim3 g((C_ + 63) / 64, BL_ / 128);
        gemm_wmma_kernel<0, 1, 0><<<g, NT, 0, stream>>>(buf6, h_outprj, nullptr, buf7,
                                                        BL_, C_, E_, E_, C_);
    }

    // 4) gated fuse + BN2 + skip residual
    gate_kernel<<<(B_ * HW_ + NT - 1) / NT, NT, 0, stream>>>(buf1, buf7, gate_w, gate_b, buf0);
    bn_stats_kernel<<<C_, NT, 0, stream>>>(buf0, stats);
    bn_apply_kernel<<<(NPIX + NT - 1) / NT, NT, 0, stream>>>(buf0, stats, bn2_g, bn2_b, skip, buf1);

    // 5) 3x3 conv as implicit GEMM (WMMA) + maxpool
    {
        dim3 g((CO_ + 63) / 64, BL_ / 128);
        gemm_wmma_kernel<1, 1, 0><<<g, NT, 0, stream>>>(buf1, h_conv3, conv3_b, buf2,
                                                        BL_, CO_, KC3_, 0, CO_);
    }
    maxpool_kernel<<<(B_ * CO_ * 28 * 28 + NT - 1) / NT, NT, 0, stream>>>(buf2, outp);
}
